// EquShiftQ2DF3P40_68281390072662
// MI455X (gfx1250) — matrix operand order repaired and checked by reading, not compiled, not run on hardware
//
#include <hip/hip_runtime.h>
#include <hip/hip_bf16.h>

typedef __attribute__((ext_vector_type(16))) __bf16 v16bf;
typedef __attribute__((ext_vector_type(2)))  __bf16 v2bf;
typedef __attribute__((ext_vector_type(8)))  float  v8f;

#define GT_M 32
#define GT_N 64
#define GT_K 32
#define LDSTRIDE (GT_K + 4)   // 36 elems = 72B row: 8B-aligned rows, bank-staggered

// ---------------------------------------------------------------------------
// bf16 WMMA GEMM: C[M,N] = act(A[M,K] * W[K,N] + bias), fp32 in/out, bf16 math,
// f32 accumulate. Block: 256 threads = 8 waves, block tile 32x64, one 16x16
// WMMA tile per wave. Global->register software pipeline: next K-tile's loads
// are issued before the current tile's WMMA, so s_wait_loadcnt lands after a
// full compute phase instead of immediately after the load.
// PRECONDITIONS (hold for every launch below): M % 32 == 0, K % 32 == 0,
// N % 4 == 0. Only N may be non-multiple of the 64-wide block tile.
// ---------------------------------------------------------------------------
__global__ __launch_bounds__(256)
void gemm_bf16_wmma(const float* __restrict__ A, int lda,
                    const float* __restrict__ Wt, int ldw,
                    float* __restrict__ C, int ldc,
                    const float* __restrict__ bias,
                    int M, int N, int K, int relu)
{
    __shared__ __bf16 As[GT_M][LDSTRIDE];   // [m][k]
    __shared__ __bf16 Bs[GT_N][LDSTRIDE];   // [n][k] (transposed: K-pairs contiguous)

    const int tid   = threadIdx.x;
    const int wave  = tid >> 5;
    const int lane  = tid & 31;
    const int m0    = blockIdx.y * GT_M;
    const int n0    = blockIdx.x * GT_N;
    const int wm    = (wave >> 2) * 16;   // 0 / 16
    const int wn    = (wave & 3)  * 16;   // 0..48
    const int rc    = lane & 15;          // row (A) / col (B,C)
    const int khalf = lane >> 4;          // K half select

    // fixed per-thread staging assignment
    const int sam = tid >> 3;             // A: m row 0..31
    const int sak = (tid & 7) * 4;        // A: k quad
    const int sbk = tid >> 3;             // B: k row 0..31
    const int sbn = (tid & 7) * 8;        // B: n octet
    const bool bok0 = (n0 + sbn)     < N; // float4-granular N guards (N%4==0)
    const bool bok1 = (n0 + sbn + 4) < N;

    const float* aptr = A  + (long)(m0 + sam) * lda + sak;
    const float* bptr = Wt + (long)sbk * ldw + n0 + sbn;
    const __bf16 z = (__bf16)0.0f;

    // register-resident next tile (software pipeline stage)
    float4 ra, rb0, rb1;
    auto load_tile = [&](int k0) {
        ra = *(const float4*)(aptr + k0);
        const float* bp = bptr + (long)k0 * ldw;
        const float* p0 = bok0 ? bp       : Wt;  // branch-free addr clamp
        const float* p1 = bok1 ? (bp + 4) : Wt;
        rb0 = *(const float4*)p0;
        rb1 = *(const float4*)p1;
    };
    load_tile(0);

    v8f acc = {};

    for (int k0 = 0; k0 < K; k0 += GT_K) {
        __syncthreads();                       // prior gathers done, LDS reusable
        // ---- convert + store current tile (regs -> LDS) ----
        {
            v2bf a0 = { (__bf16)ra.x, (__bf16)ra.y };
            v2bf a1 = { (__bf16)ra.z, (__bf16)ra.w };
            *(v2bf*)&As[sam][sak]     = a0;
            *(v2bf*)&As[sam][sak + 2] = a1;
            Bs[sbn + 0][sbk] = bok0 ? (__bf16)rb0.x : z;
            Bs[sbn + 1][sbk] = bok0 ? (__bf16)rb0.y : z;
            Bs[sbn + 2][sbk] = bok0 ? (__bf16)rb0.z : z;
            Bs[sbn + 3][sbk] = bok0 ? (__bf16)rb0.w : z;
            Bs[sbn + 4][sbk] = bok1 ? (__bf16)rb1.x : z;
            Bs[sbn + 5][sbk] = bok1 ? (__bf16)rb1.y : z;
            Bs[sbn + 6][sbk] = bok1 ? (__bf16)rb1.z : z;
            Bs[sbn + 7][sbk] = bok1 ? (__bf16)rb1.w : z;
        }
        __syncthreads();
        // ---- issue next tile's global loads before the compute phase ----
        if (k0 + GT_K < K) load_tile(k0 + GT_K);   // uniform branch

        // ---- gather fragments per ISA 7.12.2 (16-bit 16x32 A / 32x16 B) ----
        v16bf af, bf;
        #pragma unroll
        for (int v = 0; v < 8; ++v) {
            int ka = ((v & 4) << 2) + khalf * 8 + (v & 3) * 2;  // A K-pair
            v2bf ap = *(const v2bf*)&As[wm + rc][ka];
            af[2 * v]     = ap.x;
            af[2 * v + 1] = ap.y;
            int kb = ((v & 4) << 2) + khalf * 8 + (v & 3) * 2;  // B K-pair
            v2bf bp2 = *(const v2bf*)&Bs[wn + rc][kb];
            bf[2 * v]     = bp2.x;
            bf[2 * v + 1] = bp2.y;
        }
        acc = __builtin_amdgcn_wmma_f32_16x16x32_bf16(
                  false, af, false, bf, (short)0, acc, false, false);
    }

    // epilogue: bias + optional ReLU; only N needs guarding
    int gn = n0 + wn + rc;
    if (gn < N) {
        float bv = bias[gn];
        #pragma unroll
        for (int r = 0; r < 8; ++r) {
            int gm = m0 + wm + khalf * 8 + r;
            float v = acc[r] + bv;
            if (relu) v = fmaxf(v, 0.0f);
            C[(long)gm * ldc + gn] = v;
        }
    }
}

// ---------------------------------------------------------------------------
// Direct conv, fused bias + ReLU + optional 2x2 maxpool.
// bias index = co / bias_div  (bias_div=2 implements fb() channel-repeat).
// ---------------------------------------------------------------------------
__global__ __launch_bounds__(256)
void conv2d_relu_pool(const float* __restrict__ in, long in_bstride,
                      const float* __restrict__ w,
                      const float* __restrict__ bias, int bias_div,
                      float* __restrict__ out,
                      int B, int Ci, int H, int W,
                      int Co, int KH, int KW, int pad, int stride,
                      int OH, int OW, int pool)
{
    int PH = pool ? (OH >> 1) : OH;
    int PW = pool ? (OW >> 1) : OW;
    long total = (long)B * Co * PH * PW;
    long idx = (long)blockIdx.x * blockDim.x + threadIdx.x;
    if (idx >= total) return;

    int px = (int)(idx % PW); long t = idx / PW;
    int py = (int)(t % PH);   t /= PH;
    int co = (int)(t % Co);
    int b  = (int)(t / Co);

    const float* wp = w + (long)co * Ci * KH * KW;
    const float* ip = in + (long)b * in_bstride;
    float bv = bias[co / bias_div];

    int ns = pool ? 2 : 1;
    float m = -1.0f;   // post-ReLU values are >= 0
    for (int sy = 0; sy < ns; ++sy) {
        for (int sx = 0; sx < ns; ++sx) {
            int oy = pool ? py * 2 + sy : py;
            int ox = pool ? px * 2 + sx : px;
            float s = bv;
            for (int ci = 0; ci < Ci; ++ci) {
                const float* ipc = ip + (long)ci * H * W;
                const float* wpc = wp + (long)ci * KH * KW;
                for (int ky = 0; ky < KH; ++ky) {
                    int iy = oy * stride - pad + ky;
                    if (iy < 0 || iy >= H) continue;
                    for (int kx = 0; kx < KW; ++kx) {
                        int ix = ox * stride - pad + kx;
                        if (ix < 0 || ix >= W) continue;
                        s += ipc[iy * W + ix] * wpc[ky * KW + kx];
                    }
                }
            }
            s = fmaxf(s, 0.0f);
            m = fmaxf(m, s);
        }
    }
    out[idx] = m;
}

// ---------------------------------------------------------------------------
// Equivariant kernel expansion. rot90 once: out[i,j] = in[j, 2-i].
// sym: 0.5*(k[i,j] + k[2-i,2-j]).
// ---------------------------------------------------------------------------
__global__ void expand_tq_kernel(const float* __restrict__ kappa,
                                 float* __restrict__ out, int Co)
{
    int total = Co * 2 * 9;
    int e = blockIdx.x * blockDim.x + threadIdx.x;
    if (e >= total) return;
    int j = e % 3, i = (e / 3) % 3, oc2 = e / 9;
    int g = oc2 & 1, co = oc2 >> 1;
    int si = (g == 0) ? i : j;
    int sj = (g == 0) ? j : (2 - i);
    const float* kp = kappa + (long)co * 9;
    out[e] = 0.5f * (kp[si * 3 + sj] + kp[(2 - si) * 3 + (2 - sj)]);
}

__global__ void expand_qq_kernel(const float* __restrict__ kappa,
                                 float* __restrict__ out, int Co, int Ci)
{
    int total = Co * 2 * Ci * 2 * 9;
    int e = blockIdx.x * blockDim.x + threadIdx.x;
    if (e >= total) return;
    int j = e % 3; int t = e / 3;
    int i = t % 3; t /= 3;
    int ci2 = t % (Ci * 2); t /= (Ci * 2);
    int oc2 = t;
    int g = oc2 & 1, co = oc2 >> 1;
    int f = ci2 & 1, ci = ci2 >> 1;
    int fp = (f - g) & 1;                 // fiber roll
    int si = (g == 0) ? i : j;
    int sj = (g == 0) ? j : (2 - i);
    const float* kp = kappa + (((long)(co * Ci + ci) * 2 + fp) * 9);
    out[e] = 0.5f * (kp[si * 3 + sj] + kp[(2 - si) * 3 + (2 - sj)]);
}

// ---------------------------------------------------------------------------
// Dynamic-weight head: feat = relu(einsum(Wb, xf) + bias); out = feat @ W2^T + b
// One thread per batch sample (tiny: 32*32 + 4*32 MACs).
// ---------------------------------------------------------------------------
__global__ void head_kernel(const float* __restrict__ x5,     // (B,32)
                            const float* __restrict__ wb,     // (B,528)
                            const float* __restrict__ kappa2, // (2,16,2)
                            const float* __restrict__ b2f,    // (2)
                            float* __restrict__ out, int B)   // (B,4)
{
    int b = blockIdx.x * blockDim.x + threadIdx.x;
    if (b >= B) return;
    const float* xf = x5 + (long)b * 32;
    const float* w  = wb + (long)b * 528;
    float feat[32];
    for (int o = 0; o < 16; ++o)
        for (int g = 0; g < 2; ++g) {
            float s = w[512 + o];
            for (int i = 0; i < 16; ++i)
                for (int f = 0; f < 2; ++f)
                    s += w[o * 32 + i * 2 + ((f - g) & 1)] * xf[i * 2 + f];
            feat[o * 2 + g] = fmaxf(s, 0.0f);
        }
    for (int c = 0; c < 2; ++c)
        for (int g = 0; g < 2; ++g) {
            float s = b2f[c];
            for (int i = 0; i < 16; ++i)
                for (int f = 0; f < 2; ++f)
                    s += kappa2[c * 32 + i * 2 + ((f - g) & 1)] * feat[i * 2 + f];
            out[(long)b * 4 + c * 2 + g] = s;
        }
}

// ---------------------------------------------------------------------------
extern "C" void kernel_launch(void* const* d_in, const int* in_sizes, int n_in,
                              void* d_out, int out_size, void* d_ws, size_t ws_size,
                              hipStream_t stream)
{
    const float* obs   = (const float*)d_in[0];   // (256,256,8,8)
    const float* patch = (const float*)d_in[1];   // (256,2,40,40)
    const float* k1    = (const float*)d_in[2];
    const float* b1    = (const float*)d_in[3];
    const float* k2    = (const float*)d_in[4];
    const float* b2    = (const float*)d_in[5];
    const float* k3    = (const float*)d_in[6];
    const float* b3    = (const float*)d_in[7];
    const float* k4    = (const float*)d_in[8];
    const float* b4    = (const float*)d_in[9];
    const float* k5    = (const float*)d_in[10];
    const float* b5    = (const float*)d_in[11];
    const float* Wes1  = (const float*)d_in[12];
    const float* bes1  = (const float*)d_in[13];
    const float* Wes2  = (const float*)d_in[14];
    const float* bes2  = (const float*)d_in[15];
    const float* Wi1   = (const float*)d_in[16];
    const float* bi1   = (const float*)d_in[17];
    const float* Wi2   = (const float*)d_in[18];
    const float* bi2   = (const float*)d_in[19];
    const float* Wif   = (const float*)d_in[20];
    const float* bif   = (const float*)d_in[21];
    const float* Wdf   = (const float*)d_in[22];
    const float* bdf   = (const float*)d_in[23];
    const float* kap2  = (const float*)d_in[24];
    const float* b2f   = (const float*)d_in[25];
    float* out = (float*)d_out;

    const int B = 256;
    float* ws = (float*)d_ws;
    size_t o = 0;
    float* W1e  = ws + o; o += 64 * 9;
    float* W2e  = ws + o; o += 128 * 64 * 9;
    float* W3e  = ws + o; o += 256 * 128 * 9;
    float* W4e  = ws + o; o += 64 * 256 * 9;
    float* W5e  = ws + o; o += 32 * 64 * 9;
    float* es1  = ws + o; o += (size_t)B * 1024;
    float* dfin = ws + o; o += (size_t)B * 1024;   // [es2 | ihv]
    float* wbuf = ws + o; o += (size_t)B * 528;
    float* h1   = ws + o; o += (size_t)B * 32 * 400;
    float* h2   = ws + o; o += (size_t)B * 64 * 100;
    float* p1   = ws + o; o += (size_t)B * 64 * 400;
    float* p2   = ws + o; o += (size_t)B * 128 * 100;
    float* x3   = ws + o; o += (size_t)B * 256 * 64;
    float* p4   = ws + o; o += (size_t)B * 64 * 9;
    float* x5   = ws + o; o += (size_t)B * 32;
    (void)ws_size; (void)n_in; (void)in_sizes; (void)out_size;

    auto blocks = [](long n) { return (int)((n + 255) / 256); };

    // --- expand equivariant weights ---
    expand_tq_kernel<<<blocks(64 * 9), 256, 0, stream>>>(k1, W1e, 32);
    expand_qq_kernel<<<blocks(128L * 64 * 9), 256, 0, stream>>>(k2, W2e, 64, 32);
    expand_qq_kernel<<<blocks(256L * 128 * 9), 256, 0, stream>>>(k3, W3e, 128, 64);
    expand_qq_kernel<<<blocks(64L * 256 * 9), 256, 0, stream>>>(k4, W4e, 32, 128);
    expand_qq_kernel<<<blocks(32L * 64 * 9), 256, 0, stream>>>(k5, W5e, 16, 32);

    // --- equivariant conv stack (img = patch[:,0]), pool fused ---
    conv2d_relu_pool<<<blocks((long)B * 64 * 400), 256, 0, stream>>>(
        patch, 3200, W1e, b1, 2, p1, B, 1, 40, 40, 64, 3, 3, 1, 1, 40, 40, 1);
    conv2d_relu_pool<<<blocks((long)B * 128 * 100), 256, 0, stream>>>(
        p1, 64L * 400, W2e, b2, 2, p2, B, 64, 20, 20, 128, 3, 3, 1, 1, 20, 20, 1);
    conv2d_relu_pool<<<blocks((long)B * 256 * 64), 256, 0, stream>>>(
        p2, 128L * 100, W3e, b3, 2, x3, B, 128, 10, 10, 256, 3, 3, 0, 1, 8, 8, 0);
    conv2d_relu_pool<<<blocks((long)B * 64 * 9), 256, 0, stream>>>(
        x3, 256L * 64, W4e, b4, 2, p4, B, 256, 8, 8, 64, 3, 3, 0, 1, 6, 6, 1);
    conv2d_relu_pool<<<blocks((long)B * 32), 256, 0, stream>>>(
        p4, 64L * 9, W5e, b5, 2, x5, B, 64, 3, 3, 32, 3, 3, 0, 1, 1, 1, 0);

    // --- ih conv stem (ih = patch[:,1]) ---
    conv2d_relu_pool<<<blocks((long)B * 32 * 400), 256, 0, stream>>>(
        patch + 1600, 3200, Wi1, bi1, 1, h1, B, 1, 40, 40, 32, 3, 3, 1, 2, 20, 20, 0);
    conv2d_relu_pool<<<blocks((long)B * 64 * 100), 256, 0, stream>>>(
        h1, 32L * 400, Wi2, bi2, 1, h2, B, 32, 20, 20, 64, 3, 3, 1, 2, 10, 10, 0);

    // --- WMMA GEMM chain (M%32==0, K%32==0, N%4==0 for every call) ---
    dim3 blk(256);
    dim3 g1((1024 + GT_N - 1) / GT_N, (B + GT_M - 1) / GT_M);
    gemm_bf16_wmma<<<g1, blk, 0, stream>>>(obs, 16384, Wes1, 1024,
                                           es1, 1024, bes1, B, 1024, 16384, 1);
    dim3 g2((512 + GT_N - 1) / GT_N, (B + GT_M - 1) / GT_M);
    gemm_bf16_wmma<<<g2, blk, 0, stream>>>(es1, 1024, Wes2, 512,
                                           dfin, 1024, bes2, B, 512, 1024, 1);
    gemm_bf16_wmma<<<g2, blk, 0, stream>>>(h2, 6400, Wif, 512,
                                           dfin + 512, 1024, bif, B, 512, 6400, 1);
    dim3 g3((528 + GT_N - 1) / GT_N, (B + GT_M - 1) / GT_M);
    gemm_bf16_wmma<<<g3, blk, 0, stream>>>(dfin, 1024, Wdf, 528,
                                           wbuf, 528, bdf, B, 528, 1024, 0);

    // --- dynamic head ---
    head_kernel<<<1, 256, 0, stream>>>(x5, wbuf, kap2, b2f, out, B);
}
